// BertBaseline_51376398795334
// MI455X (gfx1250) — compile-verified, matrix operand order checked by reading, not run on hardware
//
#include <hip/hip_runtime.h>
#include <hip/hip_bf16.h>
#include <math.h>

// ---------------- problem constants ----------------
#define D_    512
#define H_    8
#define DH_   64
#define FF_   2048
#define T_    512
#define B_    16
#define NBL_  12
#define NI_   96          // upper bound on instructions per sequence
#define NS_   16          // max instruction length
#define L2_   (NI_*NS_)   // 1536
#define PAD_TOK 0
#define END_TOK 9

// input indices (setup_inputs dict insertion order, nested dicts in insertion order)
#define IN_X 0
#define IN_EMB 1
#define BERT_BASE 2
#define BB_BASE 14
#define SEQ_BASE 26
#define OP_BASE 38
#define IN_PW 50
#define IN_PB 51

typedef __attribute__((ext_vector_type(16))) __bf16         v16bf;
typedef __attribute__((ext_vector_type(16))) unsigned short v16u;
typedef __attribute__((ext_vector_type(8)))  unsigned short v8u;
typedef __attribute__((ext_vector_type(8)))  float          v8f;

static __device__ __forceinline__ unsigned short f2bf(float f) {
  unsigned int u = __float_as_uint(f);
  u += 0x7FFFu + ((u >> 16) & 1u);         // round-to-nearest-even
  return (unsigned short)(u >> 16);
}
static __device__ __forceinline__ unsigned int pack2(float a, float b) {
  return (unsigned int)f2bf(a) | ((unsigned int)f2bf(b) << 16);
}

// 16-bit WMMA fragment: lane<16 holds K in {kb..kb+7, kb+16..kb+23} (kb=0),
// lane>=16 kb=8 (ISA 7.12.2). With contiguous-K storage this is two 16B chunks.
static __device__ __forceinline__ v16u ld_frag(const unsigned short* p) {
  v8u lo = *(const v8u*)p;
  v8u hi = *(const v8u*)(p + 16);
  return __builtin_shufflevector(lo, hi, 0,1,2,3,4,5,6,7,8,9,10,11,12,13,14,15);
}

static __device__ __forceinline__ v8f wmma_bf16(v16u a, v16u b, v8f c) {
  return __builtin_amdgcn_wmma_f32_16x16x32_bf16(
      false, __builtin_bit_cast(v16bf, a),
      false, __builtin_bit_cast(v16bf, b),
      (short)0, c, false, false);
}

static __device__ __forceinline__ float pe_val(int pos, int dd) {
  int i2 = dd >> 1;
  float div = __expf(-logf(10000.0f) * (float)(2 * i2) / (float)D_);
  float ang = (float)pos * div;
  return (dd & 1) ? __cosf(ang) : __sinf(ang);
}

// ---------------- elementwise kernels ----------------
__global__ void k_embed(const int* __restrict__ x, const float* __restrict__ emb,
                        float* __restrict__ h) {
  long i = (long)blockIdx.x * blockDim.x + threadIdx.x;
  if (i >= (long)B_ * T_ * D_) return;
  int dd = (int)(i % D_);
  long bt = i / D_;
  int t = (int)(bt % T_);
  int tok = x[bt];
  h[i] = emb[(size_t)tok * D_ + dd] + pe_val(t, dd);
}

__global__ void k_mask_from_x(const int* __restrict__ x, unsigned char* __restrict__ m) {
  int i = blockIdx.x * blockDim.x + threadIdx.x;
  if (i < B_ * T_) m[i] = (x[i] == PAD_TOK) ? 1 : 0;
}

// wave-per-row LayerNorm over D=512 (16 elems/lane, wave32 shuffles)
__global__ void k_layernorm(const float* __restrict__ in, const float* __restrict__ g,
                            const float* __restrict__ b, float* __restrict__ out, int M) {
  int wv = (blockIdx.x * blockDim.x + threadIdx.x) >> 5;
  int lane = threadIdx.x & 31;
  if (wv >= M) return;
  const float* row = in + (size_t)wv * D_;
  float vals[16];
  float s = 0.f;
#pragma unroll
  for (int j = 0; j < 16; j++) { vals[j] = row[lane + 32 * j]; s += vals[j]; }
#pragma unroll
  for (int o = 16; o >= 1; o >>= 1) s += __shfl_xor(s, o, 32);
  float mu = s * (1.0f / D_);
  float v = 0.f;
#pragma unroll
  for (int j = 0; j < 16; j++) { float d = vals[j] - mu; v += d * d; }
#pragma unroll
  for (int o = 16; o >= 1; o >>= 1) v += __shfl_xor(v, o, 32);
  float inv = rsqrtf(v * (1.0f / D_) + 1e-5f);
  float* orow = out + (size_t)wv * D_;
#pragma unroll
  for (int j = 0; j < 16; j++) {
    int c = lane + 32 * j;
    orow[c] = (vals[j] - mu) * inv * g[c] + b[c];
  }
}

// ---------------- WMMA GEMM: C = act(A[MxK] @ W[KxN] + bias) (+resid) ----------------
// 128x128 block tile, 8 waves x (32x64), bf16 LDS staging, double-buffered.
#define BM 128
#define BN 128
#define BK 32
#define SAP 40   // padded row stride (ushorts): 80B, 16B-aligned, conflict-free
__global__ __launch_bounds__(256) void k_gemm(const float* __restrict__ A,
                                              const float* __restrict__ W,
                                              const float* __restrict__ bias,
                                              const float* __restrict__ resid,
                                              float* __restrict__ C,
                                              int M, int N, int K, int act) {
  __shared__ __align__(16) unsigned short sA[2][BM][SAP];    // A tile,  [m][k]
  __shared__ __align__(16) unsigned short sBt[2][BN][SAP];   // B tile transposed, [n][k]
  int tid = threadIdx.x;
  int wave = tid >> 5, lane = tid & 31;
  int wm = wave >> 1, wn = wave & 1;      // 4x2 waves, each 32x64 output
  int bm0 = blockIdx.y * BM, bn0 = blockIdx.x * BN;
  int am = lane & 15;
  int kb = (lane < 16) ? 0 : 8;

  // per-thread staging coordinates (vectorized float4 tile loads)
  int ar = tid >> 3, ac4 = (tid & 7) * 4;          // A: 32 rows x 8 chunks
  int br = tid >> 5, bc4 = (tid & 31) * 4;         // B: 32 rows x 32 chunks (x4 u-steps of 8 rows)
  int gra = bm0 + ar; if (gra >= M) gra = M - 1;   // clamp (out-of-range rows never stored)

  float4 ra, rb[4];
  auto load_tiles = [&](int k0) {
    ra = *(const float4*)(A + (size_t)gra * K + k0 + ac4);
#pragma unroll
    for (int u = 0; u < 4; u++)
      rb[u] = *(const float4*)(W + (size_t)(k0 + br + u * 8) * N + bn0 + bc4);
  };
  auto store_tiles = [&](int p) {
    *(uint2*)(&sA[p][ar][ac4]) = make_uint2(pack2(ra.x, ra.y), pack2(ra.z, ra.w));
#pragma unroll
    for (int u = 0; u < 4; u++) {
      int r = br + u * 8;
      sBt[p][bc4 + 0][r] = f2bf(rb[u].x);
      sBt[p][bc4 + 1][r] = f2bf(rb[u].y);
      sBt[p][bc4 + 2][r] = f2bf(rb[u].z);
      sBt[p][bc4 + 3][r] = f2bf(rb[u].w);
    }
  };

  v8f acc[2][4];
#pragma unroll
  for (int mi = 0; mi < 2; mi++)
#pragma unroll
    for (int ni = 0; ni < 4; ni++)
      acc[mi][ni] = (v8f){0.f, 0.f, 0.f, 0.f, 0.f, 0.f, 0.f, 0.f};

  load_tiles(0);
  store_tiles(0);
  int p = 0;
  for (int k0 = 0; k0 < K; k0 += BK) {
    __syncthreads();
    bool more = (k0 + BK) < K;
    if (more) load_tiles(k0 + BK);     // global loads overlap with WMMA below

    v16u au[2], bu[4];
#pragma unroll
    for (int mi = 0; mi < 2; mi++)
      au[mi] = ld_frag(&sA[p][wm * 32 + mi * 16 + am][kb]);
#pragma unroll
    for (int ni = 0; ni < 4; ni++)
      bu[ni] = ld_frag(&sBt[p][wn * 64 + ni * 16 + am][kb]);
#pragma unroll
    for (int mi = 0; mi < 2; mi++)
#pragma unroll
      for (int ni = 0; ni < 4; ni++)
        acc[mi][ni] = wmma_bf16(au[mi], bu[ni], acc[mi][ni]);

    if (more) store_tiles(p ^ 1);      // fill the other buffer
    p ^= 1;
  }

  int ch = lane >> 4, cc = lane & 15;
#pragma unroll
  for (int mi = 0; mi < 2; mi++) {
#pragma unroll
    for (int ni = 0; ni < 4; ni++) {
      int col = bn0 + wn * 64 + ni * 16 + cc;
      float bv = bias[col];
#pragma unroll
      for (int vg = 0; vg < 8; vg++) {
        int row = bm0 + wm * 32 + mi * 16 + vg + 8 * ch;
        if (row < M) {
          float xv = acc[mi][ni][vg] + bv;
          if (act == 1) xv = 0.5f * xv * (1.0f + erff(xv * 0.70710678118f));
          if (resid) xv += resid[(size_t)row * N + col];
          C[(size_t)row * N + col] = xv;
        }
      }
    }
  }
}

// ---------------- QKV split / head merge ----------------
__global__ void k_split_qkv(const float* __restrict__ wide, float* __restrict__ q,
                            float* __restrict__ k, float* __restrict__ v,
                            int nb, int Tq) {
  long i = (long)blockIdx.x * blockDim.x + threadIdx.x;
  long tot = (long)nb * Tq * D_;
  if (i >= tot) return;
  int dd = (int)(i % DH_);
  int hh = (int)((i / DH_) % H_);
  int t  = (int)((i / D_) % Tq);
  int b  = (int)(i / ((long)D_ * Tq));
  size_t src = ((size_t)b * Tq + t) * (3 * D_);
  size_t dst = (((size_t)b * H_ + hh) * Tq + t) * DH_ + dd;
  int c = hh * DH_ + dd;
  q[dst] = wide[src + c];
  k[dst] = wide[src + D_ + c];
  v[dst] = wide[src + 2 * D_ + c];
}

__global__ void k_merge_heads(const float* __restrict__ a, float* __restrict__ ctx,
                              int nb, int Tq) {
  long i = (long)blockIdx.x * blockDim.x + threadIdx.x;
  long tot = (long)nb * Tq * D_;
  if (i >= tot) return;
  int dd = (int)(i % DH_);
  int hh = (int)((i / DH_) % H_);
  int t  = (int)((i / D_) % Tq);
  int b  = (int)(i / ((long)D_ * Tq));
  ctx[((size_t)b * Tq + t) * D_ + hh * DH_ + dd] =
      a[(((size_t)b * H_ + hh) * Tq + t) * DH_ + dd];
}

// ---------------- flash attention (WMMA, online softmax, LDS-staged K/V) ----------------
// Q,K,V,O: [nb*H, Tq, 64]; mask: [nb, Tq] (1 = padded key). One wave = 16 query rows.
// All 8 waves of a block share one (batch,head); K/V tiles staged in LDS per block.
__global__ __launch_bounds__(256) void k_attn(const float* __restrict__ Q,
                                              const float* __restrict__ K,
                                              const float* __restrict__ V,
                                              const unsigned char* __restrict__ mask,
                                              float* __restrict__ O, int Tq) {
  __shared__ __align__(16) unsigned short sK[32][72];   // K tile [key][dh], padded
  __shared__ __align__(16) unsigned short sVt[64][SAP]; // V tile transposed [dh][key]
  __shared__ float sS[8][16][33];
  __shared__ __align__(16) unsigned short sP[8][16][SAP];
  __shared__ float sFac[8][16];

  int tid = threadIdx.x;
  int wave = tid >> 5, lane = tid & 31;
  int q0 = blockIdx.x * 128 + wave * 16;
  bool active = (q0 < Tq);
  int bh = blockIdx.y;
  int b = bh / H_;
  const float* Qp = Q + (size_t)bh * Tq * DH_;
  const float* Kp = K + (size_t)bh * Tq * DH_;
  const float* Vp = V + (size_t)bh * Tq * DH_;
  float* Op = O + (size_t)bh * Tq * DH_;
  const unsigned char* mp = mask + (size_t)b * Tq;

  int am = lane & 15;
  int kb = (lane < 16) ? 0 : 8;
  int ch = lane >> 4;

  // Q fragments (two k=32 chunks of dh=64), pre-scaled by 1/sqrt(dh)
  v16u qa[2];
  if (active) {
#pragma unroll
    for (int f = 0; f < 2; f++) {
      const float* qp = Qp + (size_t)(q0 + am) * DH_ + f * 32 + kb;
      float4 a0 = *(const float4*)(qp);
      float4 a1 = *(const float4*)(qp + 4);
      float4 b0 = *(const float4*)(qp + 16);
      float4 b1 = *(const float4*)(qp + 20);
      v16u t;
      t[0] = f2bf(a0.x * 0.125f); t[1] = f2bf(a0.y * 0.125f);
      t[2] = f2bf(a0.z * 0.125f); t[3] = f2bf(a0.w * 0.125f);
      t[4] = f2bf(a1.x * 0.125f); t[5] = f2bf(a1.y * 0.125f);
      t[6] = f2bf(a1.z * 0.125f); t[7] = f2bf(a1.w * 0.125f);
      t[8]  = f2bf(b0.x * 0.125f); t[9]  = f2bf(b0.y * 0.125f);
      t[10] = f2bf(b0.z * 0.125f); t[11] = f2bf(b0.w * 0.125f);
      t[12] = f2bf(b1.x * 0.125f); t[13] = f2bf(b1.y * 0.125f);
      t[14] = f2bf(b1.z * 0.125f); t[15] = f2bf(b1.w * 0.125f);
      qa[f] = t;
    }
  }

  v8f acc[4];
#pragma unroll
  for (int nt = 0; nt < 4; nt++) acc[nt] = (v8f){0.f,0.f,0.f,0.f,0.f,0.f,0.f,0.f};
  float mrow = -3e38f, lrow = 0.f;   // valid in lanes 0..15 (row = lane)

  for (int kt = 0; kt < Tq; kt += 32) {
    __syncthreads();
    // stage K (row-major) and V (transposed) tiles: 32 keys x 64 dh, float4 loads
#pragma unroll
    for (int u = 0; u < 2; u++) {
      int i = tid + u * 256;
      int r = i >> 4, c4 = (i & 15) * 4;
      int key = kt + r;
      if (key >= Tq) key = Tq - 1;
      float4 kv = *(const float4*)(Kp + (size_t)key * DH_ + c4);
      float4 vv = *(const float4*)(Vp + (size_t)key * DH_ + c4);
      *(uint2*)(&sK[r][c4]) = make_uint2(pack2(kv.x, kv.y), pack2(kv.z, kv.w));
      sVt[c4 + 0][r] = f2bf(vv.x);
      sVt[c4 + 1][r] = f2bf(vv.y);
      sVt[c4 + 2][r] = f2bf(vv.z);
      sVt[c4 + 3][r] = f2bf(vv.w);
    }
    __syncthreads();
    if (!active) continue;

    // S tile = Q · K^T for 32 keys (two 16-key subtiles, k=64 contraction)
#pragma unroll
    for (int nt = 0; nt < 2; nt++) {
      v8f c = (v8f){0.f,0.f,0.f,0.f,0.f,0.f,0.f,0.f};
#pragma unroll
      for (int f = 0; f < 2; f++) {
        v16u bt = ld_frag(&sK[nt * 16 + am][f * 32 + kb]);
        c = wmma_bf16(qa[f], bt, c);
      }
#pragma unroll
      for (int vg = 0; vg < 8; vg++) sS[wave][vg + 8 * ch][nt * 16 + am] = c[vg];
    }
    __builtin_amdgcn_wave_barrier();

    // online softmax: lanes 0..15 each own one query row
    if (lane < 16) {
      float tm = -3e38f;
      for (int c2 = 0; c2 < 32; c2++) {
        int key = kt + c2;
        float sv = (key < Tq && !mp[key]) ? sS[wave][lane][c2] : -1e30f;
        sS[wave][lane][c2] = sv;
        tm = fmaxf(tm, sv);
      }
      float nm = fmaxf(mrow, tm);
      float fac = __expf(mrow - nm);
      float ts = 0.f;
      for (int c2 = 0; c2 < 32; c2++) {
        float e = __expf(sS[wave][lane][c2] - nm);
        ts += e;
        sP[wave][lane][c2] = f2bf(e);
      }
      lrow = lrow * fac + ts;
      mrow = nm;
      sFac[wave][lane] = fac;
    }
    __builtin_amdgcn_wave_barrier();

    float fr[8];
#pragma unroll
    for (int vg = 0; vg < 8; vg++) fr[vg] = sFac[wave][vg + 8 * ch];
#pragma unroll
    for (int nt = 0; nt < 4; nt++)
#pragma unroll
      for (int vg = 0; vg < 8; vg++) acc[nt][vg] *= fr[vg];

    // P (16x32) as A fragment; P · V into 4 output subtiles of dh
    v16u pa = ld_frag(&sP[wave][am][kb]);
#pragma unroll
    for (int nt = 0; nt < 4; nt++) {
      v16u vb = ld_frag(&sVt[nt * 16 + am][kb]);
      acc[nt] = wmma_bf16(pa, vb, acc[nt]);
    }
  }

  if (!active) return;
  if (lane < 16) sFac[wave][lane] = 1.0f / lrow;
  __builtin_amdgcn_wave_barrier();
#pragma unroll
  for (int nt = 0; nt < 4; nt++)
#pragma unroll
    for (int vg = 0; vg < 8; vg++) {
      int r = vg + 8 * ch;
      Op[(size_t)(q0 + r) * DH_ + nt * 16 + am] = acc[nt][vg] * sFac[wave][r];
    }
}

// ---------------- DeepPM head helpers ----------------
__global__ void k_regroup(const int* __restrict__ x, int* __restrict__ idx,
                          unsigned char* __restrict__ tokmask,
                          unsigned char* __restrict__ opm) {
  int b = threadIdx.x;
  if (b >= B_) return;
  for (int i = 0; i < NI_ * NS_; i++) { idx[b * NI_ * NS_ + i] = 0; tokmask[b * NI_ * NS_ + i] = 1; }
  for (int i = 0; i < NI_; i++) opm[b * NI_ + i] = 1;
  int ins = 0, st = 0;
  for (int t = 0; t < T_; t++) {
    if (x[b * T_ + t] == END_TOK) {
      int ln = t - st + 1;
      if (ln > NS_) ln = NS_;
      for (int s = 0; s < ln; s++) {
        idx[(b * NI_ + ins) * NS_ + s] = st + s;
        tokmask[(b * NI_ + ins) * NS_ + s] = 0;
      }
      opm[b * NI_ + ins] = 0;
      ins++; st = t + 1;
      if (ins >= NI_) break;
    }
  }
}

__global__ void k_gather(const float* __restrict__ bert, const int* __restrict__ idx,
                         const unsigned char* __restrict__ tokmask, float* __restrict__ h) {
  long i = (long)blockIdx.x * blockDim.x + threadIdx.x;
  long tot = (long)B_ * L2_ * D_;
  if (i >= tot) return;
  int dd = (int)(i % D_);
  long row = i / D_;                  // b*L2 + ii*NS + s
  int b = (int)(row / L2_);
  int fl = (int)(row % L2_);
  h[i] = tokmask[b * L2_ + fl] ? 0.f
       : bert[((size_t)b * T_ + idx[b * L2_ + fl]) * D_ + dd];
}

__global__ void k_zero_mask(float* __restrict__ h, const unsigned char* __restrict__ m, long M) {
  long i = (long)blockIdx.x * blockDim.x + threadIdx.x;
  if (i >= M * D_) return;
  if (m[i / D_]) h[i] = 0.f;
}

__global__ void k_sum_pe(const float* __restrict__ h, float* __restrict__ hs) {
  long i = (long)blockIdx.x * blockDim.x + threadIdx.x;
  long tot = (long)B_ * NI_ * D_;
  if (i >= tot) return;
  int dd = (int)(i % D_);
  long r = i / D_;                    // b*NI + ii
  int ii = (int)(r % NI_);
  float s = 0.f;
  const float* src = h + (size_t)r * NS_ * D_ + dd;
#pragma unroll
  for (int s_ = 0; s_ < NS_; s_++) s += src[(size_t)s_ * D_];
  hs[i] = s + pe_val(ii, dd);
}

__global__ void k_final(const float* __restrict__ h, const unsigned char* __restrict__ opm,
                        const float* __restrict__ pw, const float* __restrict__ pb,
                        float* __restrict__ out) {
  __shared__ float red[256];
  int b = blockIdx.x, tid = threadIdx.x;
  float acc = 0.f;
  for (int dd = tid; dd < D_; dd += 256) {
    float s = 0.f;
    for (int ii = 0; ii < NI_; ii++)
      if (!opm[b * NI_ + ii]) s += h[((size_t)b * NI_ + ii) * D_ + dd];
    acc += s * pw[dd];
  }
  red[tid] = acc;
  __syncthreads();
  for (int o = 128; o >= 1; o >>= 1) {
    if (tid < o) red[tid] += red[tid + o];
    __syncthreads();
  }
  if (tid == 0) out[b] = red[0] + pb[0];
}

// ---------------- host orchestration ----------------
struct LayerP {
  const float *qkv_w,*qkv_b,*out_w,*out_b,*ln1_g,*ln1_b,*ln2_g,*ln2_b,*ff1_w,*ff1_b,*ff2_w,*ff2_b;
};
static LayerP layer_at(void* const* din, int base, int l) {
  LayerP p;
  p.qkv_w = (const float*)din[base+0]  + (size_t)l * D_ * 3 * D_;
  p.qkv_b = (const float*)din[base+1]  + (size_t)l * 3 * D_;
  p.out_w = (const float*)din[base+2]  + (size_t)l * D_ * D_;
  p.out_b = (const float*)din[base+3]  + (size_t)l * D_;
  p.ln1_g = (const float*)din[base+4]  + (size_t)l * D_;
  p.ln1_b = (const float*)din[base+5]  + (size_t)l * D_;
  p.ln2_g = (const float*)din[base+6]  + (size_t)l * D_;
  p.ln2_b = (const float*)din[base+7]  + (size_t)l * D_;
  p.ff1_w = (const float*)din[base+8]  + (size_t)l * D_ * FF_;
  p.ff1_b = (const float*)din[base+9]  + (size_t)l * FF_;
  p.ff2_w = (const float*)din[base+10] + (size_t)l * FF_ * D_;
  p.ff2_b = (const float*)din[base+11] + (size_t)l * D_;
  return p;
}

static inline long cdivl(long a, long b) { return (a + b - 1) / b; }

static void gemm(hipStream_t s, const float* A, const float* W, const float* bias,
                 const float* resid, float* C, int M, int N, int K, int act) {
  dim3 g((unsigned)cdivl(N, BN), (unsigned)cdivl(M, BM));
  k_gemm<<<g, 256, 0, s>>>(A, W, bias, resid, C, M, N, K, act);
}
static void lnorm(hipStream_t s, const float* in, const float* g, const float* b,
                  float* out, int M) {
  k_layernorm<<<(unsigned)cdivl((long)M * 32, 256), 256, 0, s>>>(in, g, b, out, M);
}
static void attn(hipStream_t s, const float* Q, const float* K, const float* V,
                 const unsigned char* m, float* O, int nb, int Tq) {
  dim3 g((unsigned)cdivl(Tq, 128), (unsigned)(nb * H_));
  k_attn<<<g, 256, 0, s>>>(Q, K, V, m, O, Tq);
}

static void encoder(hipStream_t s, float* h, int nb, int Tq, const unsigned char* mask,
                    void* const* din, int base, int nl, bool preLN,
                    float* bufA, float* wide, float* q, float* k, float* v,
                    float* attnb, float* ctx) {
  int M = nb * Tq;
  long nEl = (long)M * D_;
  for (int l = 0; l < nl; l++) {
    LayerP p = layer_at(din, base, l);
    if (preLN) {
      lnorm(s, h, p.ln1_g, p.ln1_b, bufA, M);
      gemm(s, bufA, p.qkv_w, p.qkv_b, nullptr, wide, M, 3 * D_, D_, 0);
      k_split_qkv<<<(unsigned)cdivl(nEl, 256), 256, 0, s>>>(wide, q, k, v, nb, Tq);
      attn(s, q, k, v, mask, attnb, nb, Tq);
      k_merge_heads<<<(unsigned)cdivl(nEl, 256), 256, 0, s>>>(attnb, ctx, nb, Tq);
      gemm(s, ctx, p.out_w, p.out_b, h, h, M, D_, D_, 0);
      lnorm(s, h, p.ln2_g, p.ln2_b, bufA, M);
      gemm(s, bufA, p.ff1_w, p.ff1_b, nullptr, wide, M, FF_, D_, 1);
      gemm(s, wide, p.ff2_w, p.ff2_b, h, h, M, D_, FF_, 0);
    } else {
      gemm(s, h, p.qkv_w, p.qkv_b, nullptr, wide, M, 3 * D_, D_, 0);
      k_split_qkv<<<(unsigned)cdivl(nEl, 256), 256, 0, s>>>(wide, q, k, v, nb, Tq);
      attn(s, q, k, v, mask, attnb, nb, Tq);
      k_merge_heads<<<(unsigned)cdivl(nEl, 256), 256, 0, s>>>(attnb, ctx, nb, Tq);
      gemm(s, ctx, p.out_w, p.out_b, h, bufA, M, D_, D_, 0);
      lnorm(s, bufA, p.ln1_g, p.ln1_b, h, M);
      gemm(s, h, p.ff1_w, p.ff1_b, nullptr, wide, M, FF_, D_, 1);
      gemm(s, wide, p.ff2_w, p.ff2_b, h, bufA, M, D_, FF_, 0);
      lnorm(s, bufA, p.ln2_g, p.ln2_b, h, M);
    }
  }
}

extern "C" void kernel_launch(void* const* d_in, const int* in_sizes, int n_in,
                              void* d_out, int out_size, void* d_ws, size_t ws_size,
                              hipStream_t stream) {
  (void)in_sizes; (void)n_in; (void)out_size; (void)ws_size;
  const int* x = (const int*)d_in[IN_X];
  const float* emb = (const float*)d_in[IN_EMB];
  const float* pw = (const float*)d_in[IN_PW];
  const float* pb = (const float*)d_in[IN_PB];
  float* out = (float*)d_out;

  size_t off = 0;
  auto wsa = [&](size_t bytes) -> void* {
    void* p = (char*)d_ws + off;
    off += (bytes + 255) & ~(size_t)255;
    return p;
  };
  const size_t MMAX = (size_t)B_ * L2_;   // 24576 rows (head max)
  float* hbert = (float*)wsa((size_t)B_ * T_ * D_ * 4);
  float* hh    = (float*)wsa(MMAX * D_ * 4);
  float* bufA  = (float*)wsa(MMAX * D_ * 4);
  float* wide  = (float*)wsa(MMAX * FF_ * 4);
  float* q     = (float*)wsa(MMAX * D_ * 4);
  float* k     = (float*)wsa(MMAX * D_ * 4);
  float* v     = (float*)wsa(MMAX * D_ * 4);
  float* attnb = (float*)wsa(MMAX * D_ * 4);
  float* ctx   = (float*)wsa(MMAX * D_ * 4);
  float* hsum  = (float*)wsa((size_t)B_ * NI_ * D_ * 4);
  int* idx            = (int*)wsa((size_t)B_ * L2_ * 4);
  unsigned char* tokm = (unsigned char*)wsa((size_t)B_ * L2_);
  unsigned char* opm  = (unsigned char*)wsa((size_t)B_ * NI_);
  unsigned char* kpmx = (unsigned char*)wsa((size_t)B_ * T_);

  // BERT: embed + 12 pre-LN layers
  k_embed<<<(unsigned)cdivl((long)B_ * T_ * D_, 256), 256, 0, stream>>>(x, emb, hbert);
  k_mask_from_x<<<(unsigned)cdivl((long)B_ * T_, 256), 256, 0, stream>>>(x, kpmx);
  encoder(stream, hbert, B_, T_, kpmx, d_in, BERT_BASE, NBL_, true,
          bufA, wide, q, k, v, attnb, ctx);

  // DeepPM head: regroup + gather
  k_regroup<<<1, B_, 0, stream>>>(x, idx, tokm, opm);
  k_gather<<<(unsigned)cdivl((long)B_ * L2_ * D_, 256), 256, 0, stream>>>(hbert, idx, tokm, hh);

  // basic-block encoder (post-LN, 2 layers) over [B, NI*NS]
  encoder(stream, hh, B_, L2_, tokm, d_in, BB_BASE, 2, false,
          bufA, wide, q, k, v, attnb, ctx);
  k_zero_mask<<<(unsigned)cdivl((long)MMAX * D_, 256), 256, 0, stream>>>(hh, tokm, (long)MMAX);

  // per-instruction encoder (post-LN, 2 layers) over [B*NI, NS]
  encoder(stream, hh, B_ * NI_, NS_, tokm, d_in, SEQ_BASE, 2, false,
          bufA, wide, q, k, v, attnb, ctx);
  k_zero_mask<<<(unsigned)cdivl((long)MMAX * D_, 256), 256, 0, stream>>>(hh, tokm, (long)MMAX);

  // sum tokens per instruction + positional enc
  k_sum_pe<<<(unsigned)cdivl((long)B_ * NI_ * D_, 256), 256, 0, stream>>>(hh, hsum);

  // op encoder (post-LN, 4 layers) over [B, NI]
  encoder(stream, hsum, B_, NI_, opm, d_in, OP_BASE, 4, false,
          bufA, wide, q, k, v, attnb, ctx);

  // masked sum over instructions + linear head
  k_final<<<B_, 256, 0, stream>>>(hsum, opm, pw, pb, out);
}
